// WaveNet_15831249453801
// MI455X (gfx1250) — compile-verified, matrix operand order checked by reading, not run on hardware
//
#include <hip/hip_runtime.h>
#include <hip/hip_bf16.h>

// ---------------------------------------------------------------------------
// WaveNet forward on MI455X (gfx1250), wave32 + WMMA f32_16x16x32_f16 + TDM.
//
// Layouts (all in d_ws):
//   resA/resB : [B][T][64]   f16  (time-major -> WMMA A-frags load contiguous)
//   skip      : [B][256][T]  f32  (channel-major -> v8f vector RMW per frag)
//   condp     : [20][B][128][128] f32 (cond conv collapsed to 128 frames)
//   condf     : [B][128][256] f32
//   packed W  : fragment-swizzled f16: elem((kc,nt),lane,e) = W[K(lane,e)][n]
//               with K(l,e) = (e<8?0:16) + (l<16?0:8) + (e&7), n = nt*16+l%16
//               -> each lane loads its 16-half B fragment as one 32B read.
//   Per-layer packed block (wc0|wc1|wr|ws) is DMA'd to LDS by the TDM once
//   per workgroup (tensor_load_to_lds + s_wait_tensorcnt).
// Workspace needed: ~167 MiB (~175 MB).
// ---------------------------------------------------------------------------

#define B_   4
#define T_   32768
#define RC_  64
#define SC_  256
#define NL_  20

// per-layer packed weight block, in halfs: wc0 | wc1 | wr | ws
#define WC0_OFF_  0
#define WC1_OFF_  8192
#define WR_OFF_   16384
#define WS_OFF_   20480
#define WL_HALF_  36864          // 73728 bytes per layer

typedef __attribute__((ext_vector_type(16))) _Float16     v16h;
typedef __attribute__((ext_vector_type(8)))  _Float16     v8h;
typedef __attribute__((ext_vector_type(8)))  float        v8f;
typedef __attribute__((ext_vector_type(4)))  unsigned int u32x4;
typedef __attribute__((ext_vector_type(8)))  int          i32x8;
typedef __attribute__((ext_vector_type(4)))  int          i32x4;

__device__ __forceinline__ v8f wmma_f16(v16h a, v16h b, v8f c) {
  return __builtin_amdgcn_wmma_f32_16x16x32_f16(
      /*neg_a=*/false, a, /*neg_b=*/false, b,
      /*c_mod=*/(short)0, c, /*reuse_a=*/false, /*reuse_b=*/false);
}

__device__ __forceinline__ v16h frag_zero() {
  v16h z;
#pragma unroll
  for (int i = 0; i < 16; ++i) z[i] = (_Float16)0.0f;
  return z;
}

// A fragment (16 time-rows x 32 channels) from a row-major [row][rowStride] f16
// array. Lane l: row = rowBase + (l&15); halves 0..7 = cols col0..col0+7,
// halves 8..15 = cols col0+16..col0+23, where col0 = kc*32 + (l>=16 ? 8 : 0).
__device__ __forceinline__ v16h loadA_rm(const _Float16* __restrict__ base,
                                         int row, int rowStride, int col0) {
  union { v16h v; v8h h[2]; } u;
  const v8h* p = reinterpret_cast<const v8h*>(base + (size_t)row * rowStride + col0);
  u.h[0] = p[0];
  u.h[1] = p[2];
  return u.v;
}

// B fragment from pre-swizzled packed weights (32B contiguous per lane).
__device__ __forceinline__ v16h loadB_pk(const _Float16* wp, int kc, int NT,
                                         int nt, int lane) {
  return *reinterpret_cast<const v16h*>(
      wp + ((((size_t)kc * NT + nt) * 32 + lane) << 4));
}

// ---------------------------------------------------------------------------
// Weight pack: src f32 [Cout][Cin][ntap] -> frag-swizzled f16 (one tap).
// ---------------------------------------------------------------------------
__global__ void pack_weight(const float* __restrict__ src, _Float16* __restrict__ dst,
                            int Cout, int Cin, int tap, int ntap) {
  int i = blockIdx.x * blockDim.x + threadIdx.x;
  int total = Cin * Cout;
  if (i >= total) return;
  int NT = Cout >> 4;
  int e  = i & 15;
  int l  = (i >> 4) & 31;
  int r  = i >> 9;
  int nt = r % NT;
  int kc = r / NT;
  int K  = ((e < 8) ? 0 : 16) + ((l < 16) ? 0 : 8) + (e & 7);
  int n  = nt * 16 + (l & 15);
  int ci = kc * 32 + K;
  dst[i] = (_Float16)src[((size_t)n * Cin + ci) * ntap + tap];
}

// ---------------------------------------------------------------------------
// Cond projection: only 128 distinct upsampled frames exist (repeat 257).
// out[b][f][oc] = bias[oc] + sum_k w[oc][k] * c[b][k][f]
// ---------------------------------------------------------------------------
__global__ void cond_proj(const float* __restrict__ c, const float* __restrict__ w,
                          const float* __restrict__ bias, float* __restrict__ out,
                          int OC) {
  int i = blockIdx.x * blockDim.x + threadIdx.x;
  int total = B_ * 128 * OC;
  if (i >= total) return;
  int oc = i % OC;
  int f  = (i / OC) & 127;
  int b  = i / (OC * 128);
  float s = bias[oc];
  const float* cp = c + (size_t)b * 64 * 128 + f;
  const float* wp = w + (size_t)oc * 64;
#pragma unroll 8
  for (int k = 0; k < 64; ++k) s += wp[k] * cp[k * 128];
  out[i] = s;
}

// ---------------------------------------------------------------------------
// First causal conv (in_ch=1, K=2) after shift_right:
// res[b][t][c] = fw[c][0]*x[t-2] + fw[c][1]*x[t-1] + fb[c]   (OOB -> 0)
// ---------------------------------------------------------------------------
__global__ void first_conv(const float* __restrict__ x, const float* __restrict__ fw,
                           const float* __restrict__ fb, _Float16* __restrict__ res) {
  int i = blockIdx.x * blockDim.x + threadIdx.x;   // (b, t) flat
  if (i >= B_ * T_) return;
  int t = i & (T_ - 1);
  float cur  = (t >= 1) ? x[i - 1] : 0.f;
  float prev = (t >= 2) ? x[i - 2] : 0.f;
  _Float16* r = res + (size_t)i * RC_;
#pragma unroll 8
  for (int ch = 0; ch < RC_; ++ch)
    r[ch] = (_Float16)(fw[ch * 2] * prev + fw[ch * 2 + 1] * cur + fb[ch]);
}

// ---------------------------------------------------------------------------
// skip init (skip layout [B][SC][T]):
// skip[b][sc][t] = skipc_b[sc] + skipc_w[sc][:] . res0[b][t][:]
// ---------------------------------------------------------------------------
__global__ __launch_bounds__(256) void skip_init(const _Float16* __restrict__ res,
                                                 const float* __restrict__ w,
                                                 const float* __restrict__ bias,
                                                 float* __restrict__ skip) {
  __shared__ _Float16 rt[64 * 64];
  int b = blockIdx.y, t0 = blockIdx.x * 64;
  const _Float16* rp = res + ((size_t)b * T_ + t0) * RC_;
  for (int i = threadIdx.x; i < 64 * 64; i += 256) rt[i] = rp[i];
  __syncthreads();
  int sc = threadIdx.x;
  float wr[64];
#pragma unroll 8
  for (int k = 0; k < 64; ++k) wr[k] = w[(size_t)sc * 64 + k];
  float bb = bias[sc];
  float* sp = skip + ((size_t)b * SC_ + sc) * T_ + t0;
  for (int tt = 0; tt < 64; tt += 4) {
    float s0 = bb, s1 = bb, s2 = bb, s3 = bb;
#pragma unroll 8
    for (int k = 0; k < 64; ++k) {
      float wk = wr[k];
      s0 += wk * (float)rt[(tt + 0) * 64 + k];
      s1 += wk * (float)rt[(tt + 1) * 64 + k];
      s2 += wk * (float)rt[(tt + 2) * 64 + k];
      s3 += wk * (float)rt[(tt + 3) * 64 + k];
    }
    float4 v; v.x = s0; v.y = s1; v.z = s2; v.w = s3;
    *reinterpret_cast<float4*>(sp + tt) = v;
  }
}

// ---------------------------------------------------------------------------
// Fused WaveNet layer. Block = 8 waves, 128 timesteps; wave w owns time rows
// [16w,16w+16). Packed weights DMA'd to LDS by the TDM, B-frags read via DS.
// 576 v_wmma per block.
// ---------------------------------------------------------------------------
__global__ __launch_bounds__(256) void wavenet_layer(
    const _Float16* __restrict__ resIn, _Float16* __restrict__ resOut,
    float* __restrict__ skip, const _Float16* __restrict__ wAll,
    const float* __restrict__ cb, const float* __restrict__ rb,
    const float* __restrict__ sb, const float* __restrict__ condp, int dil) {
  __shared__ __align__(64) _Float16 wLs[WL_HALF_];   // 72 KB packed weights
  __shared__ _Float16 hhL[128 * 72];                 // 18 KB gated hidden

  int b    = blockIdx.y;
  int t0   = blockIdx.x * 128;
  int lane = threadIdx.x & 31;
  int wv   = threadIdx.x >> 5;   // 0..7
  int hi   = lane >> 4;
  int ln   = lane & 15;
  int tw   = wv * 16;
  int tb   = t0 + tw + 8 * hi;   // base time of this lane's C/D elements

  const _Float16* rbase = resIn + (size_t)b * T_ * RC_;
  {
    int pr = t0 - dil; if (pr < 0) pr = 0;
    __builtin_prefetch(rbase + (size_t)pr * RC_, 0, 3);
  }

  // ---- stage packed weights into LDS via Tensor Data Mover -------------
#if __has_builtin(__builtin_amdgcn_tensor_load_to_lds)
  if (wv == 0) {
    unsigned long long ga = (unsigned long long)(uintptr_t)wAll;
    unsigned ldsAddr = (unsigned)(uintptr_t)&wLs[0];
    const unsigned ND = (unsigned)(WL_HALF_ * 2 / 4);   // 18432 dwords
    u32x4 g0;
    g0[0] = 1u;                                         // count=1 valid D#
    g0[1] = ldsAddr;                                    // lds_addr
    g0[2] = (unsigned)ga;                               // global_addr[31:0]
    g0[3] = (unsigned)((ga >> 32) & 0x1FFFFFFu) | (2u << 30);  // [56:32]|type=2
    i32x8 g1;
    g1[0] = (int)(2u << 16);                 // data_size = 4B, no multicast
    g1[1] = (int)((ND & 0xFFFFu) << 16);     // tensor_dim0[15:0]
    g1[2] = (int)(1u << 16);                 // tensor_dim0[31:16]=0, dim1=1
    g1[3] = (int)((ND & 0xFFFFu) << 16);     // tile_dim0 = ND
    g1[4] = 0;                               // tile_dim1/2 = 0 (unused)
    g1[5] = (int)ND;                         // tensor_dim0_stride
    g1[6] = 0;
    g1[7] = 0;
    i32x4 gz; gz[0] = 0; gz[1] = 0; gz[2] = 0; gz[3] = 0;
    i32x8 gz8;
#pragma unroll
    for (int i = 0; i < 8; ++i) gz8[i] = 0;
    __builtin_amdgcn_tensor_load_to_lds(g0, g1, gz, gz, gz8, 0);
    __builtin_amdgcn_s_wait_tensorcnt(0);
  }
#else
  for (int i = threadIdx.x; i < (int)(WL_HALF_ / 2); i += 256)
    ((unsigned*)wLs)[i] = ((const unsigned*)wAll)[i];
#endif
  __syncthreads();

  const _Float16* wc0 = &wLs[WC0_OFF_];
  const _Float16* wc1 = &wLs[WC1_OFF_];
  const _Float16* wrp = &wLs[WR_OFF_];
  const _Float16* wsp = &wLs[WS_OFF_];

  // ---- causal conv: h[128 t][128 ch] -----------------------------------
  v8f acc[8];
#pragma unroll
  for (int nt = 0; nt < 8; ++nt)
#pragma unroll
    for (int r = 0; r < 8; ++r) acc[nt][r] = 0.f;

#pragma unroll
  for (int kc = 0; kc < 2; ++kc) {
    int col0 = kc * 32 + hi * 8;
    int rowc = t0 + tw + ln;
    v16h a1 = loadA_rm(rbase, rowc, RC_, col0);            // tap1: res[t]
    v16h a0 = frag_zero();                                  // tap0: res[t-dil]
    if (rowc - dil >= 0) a0 = loadA_rm(rbase, rowc - dil, RC_, col0);
#pragma unroll
    for (int nt = 0; nt < 8; ++nt) {
      acc[nt] = wmma_f16(a0, loadB_pk(wc0, kc, 8, nt, lane), acc[nt]);
      acc[nt] = wmma_f16(a1, loadB_pk(wc1, kc, 8, nt, lane), acc[nt]);
    }
  }

  // ---- bias + cond (<=2 distinct frames per block), gate ---------------
  const float* cpb = condp + (size_t)b * 128 * 128;
  int fA = t0 / 257, fB = (t0 + 127) / 257, tSw = (fA + 1) * 257;
#pragma unroll
  for (int nt = 0; nt < 8; ++nt) {
    int ch = nt * 16 + ln;
    float bias = cb[ch];
    float cvA = cpb[fA * 128 + ch];
    float cvB = cpb[fB * 128 + ch];
#pragma unroll
    for (int r = 0; r < 8; ++r) {
      int t = tb + r;
      acc[nt][r] += bias + ((t >= tSw) ? cvB : cvA);
    }
  }
  float hh[4][8];
#pragma unroll
  for (int nt = 0; nt < 4; ++nt)
#pragma unroll
    for (int r = 0; r < 8; ++r) {
      float g = acc[nt][r], o = acc[nt + 4][r];
      float sg = 1.f / (1.f + __expf(-g));
      float th = 2.f / (1.f + __expf(-2.f * o)) - 1.f;   // tanh(o)
      hh[nt][r] = sg * th;
    }

  // ---- stage gated hidden to LDS (time-major [t][64], stride 72) -------
#pragma unroll
  for (int nt = 0; nt < 4; ++nt) {
    int ch = nt * 16 + ln;
#pragma unroll
    for (int r = 0; r < 8; ++r)
      hhL[(tw + r + 8 * hi) * 72 + ch] = (_Float16)hh[nt][r];
  }
  __syncthreads();

  // ---- re-fragment hidden as A for the 1x1 GEMMs -----------------------
  v16h ha[2];
#pragma unroll
  for (int kc = 0; kc < 2; ++kc) {
    union { v16h v; v8h h[2]; } u;
    const v8h* p = reinterpret_cast<const v8h*>(&hhL[(tw + ln) * 72 + kc * 32 + hi * 8]);
    u.h[0] = p[0]; u.h[1] = p[2];
    ha[kc] = u.v;
  }

  // ---- residual update: resOut = resIn + Wr @ hh + rb ------------------
  const _Float16* riB = resIn  + (size_t)b * T_ * RC_;
  _Float16*       roB = resOut + (size_t)b * T_ * RC_;
#pragma unroll
  for (int nt = 0; nt < 4; ++nt) {
    int ch = nt * 16 + ln;
    float rbias = rb[ch];
    v8f c;
#pragma unroll
    for (int r = 0; r < 8; ++r)
      c[r] = (float)riB[(size_t)(tb + r) * RC_ + ch] + rbias;
    c = wmma_f16(ha[0], loadB_pk(wrp, 0, 4, nt, lane), c);
    c = wmma_f16(ha[1], loadB_pk(wrp, 1, 4, nt, lane), c);
#pragma unroll
    for (int r = 0; r < 8; ++r)
      roB[(size_t)(tb + r) * RC_ + ch] = (_Float16)c[r];
  }

  // ---- skip accumulate: skip[b][sc][t] += Ws @ hh + sb  (vector RMW) ---
  float* skB = skip + (size_t)b * SC_ * T_;
#pragma unroll
  for (int nt = 0; nt < 16; ++nt) {
    int sc = nt * 16 + ln;
    float sbias = sb[sc];
    float* p = skB + (size_t)sc * T_ + tb;
    v8f c = *reinterpret_cast<const v8f*>(p);
#pragma unroll
    for (int r = 0; r < 8; ++r) c[r] += sbias;
    c = wmma_f16(ha[0], loadB_pk(wsp, 0, 16, nt, lane), c);
    c = wmma_f16(ha[1], loadB_pk(wsp, 1, 16, nt, lane), c);
    *reinterpret_cast<v8f*>(p) = c;
  }
}

// ---------------------------------------------------------------------------
// Head: relu(skip) -> fc[256x256] -> +cond -> relu -> logits[256x256].
// Block = 4 waves, 64 timesteps, LDS-staged f16 activations, 256 WMMA/wave.
// ---------------------------------------------------------------------------
__global__ __launch_bounds__(128) void wavenet_head(
    const float* __restrict__ skip, const _Float16* __restrict__ fcP,
    const _Float16* __restrict__ loP, const float* __restrict__ fcb,
    const float* __restrict__ lob, const float* __restrict__ condf,
    float* __restrict__ out) {
  __shared__ _Float16 sL[64 * 264];
  __shared__ _Float16 yL[64 * 264];

  int b = blockIdx.y, t0 = blockIdx.x * 64;
  int lane = threadIdx.x & 31;
  int wv = threadIdx.x >> 5;   // 0..3
  int hi = lane >> 4, ln = lane & 15, tw = wv * 16;
  int tb = t0 + tw + 8 * hi;

  // stage relu(skip) tile; skip is [B][SC][T] -> consecutive tids read
  // consecutive t (coalesced)
  const float* skB = skip + (size_t)b * SC_ * T_;
  for (int i = threadIdx.x; i < 64 * 256; i += 128) {
    int tt = i & 63, sc = i >> 6;
    float v = skB[(size_t)sc * T_ + t0 + tt];
    sL[tt * 264 + sc] = (_Float16)(v > 0.f ? v : 0.f);
  }
  __syncthreads();

  const float* cfb = condf + (size_t)b * 128 * 256;
  int fA = t0 / 257, fB = (t0 + 63) / 257, tSw = (fA + 1) * 257;

  v16h a[8];
#pragma unroll
  for (int kc = 0; kc < 8; ++kc) {
    union { v16h v; v8h h[2]; } u;
    const v8h* p = reinterpret_cast<const v8h*>(&sL[(tw + ln) * 264 + kc * 32 + hi * 8]);
    u.h[0] = p[0]; u.h[1] = p[2];
    a[kc] = u.v;
  }
  // fc + cond + relu -> yL
#pragma unroll
  for (int nt = 0; nt < 16; ++nt) {
    int o = nt * 16 + ln;
    v8f c;
#pragma unroll
    for (int r = 0; r < 8; ++r) c[r] = fcb[o];
#pragma unroll
    for (int kc = 0; kc < 8; ++kc) c = wmma_f16(a[kc], loadB_pk(fcP, kc, 16, nt, lane), c);
    float cvA = cfb[fA * 256 + o], cvB = cfb[fB * 256 + o];
#pragma unroll
    for (int r = 0; r < 8; ++r) {
      int t = tb + r;
      float v = c[r] + ((t >= tSw) ? cvB : cvA);
      yL[(tw + r + 8 * hi) * 264 + o] = (_Float16)(v > 0.f ? v : 0.f);
    }
  }
  __syncthreads();
#pragma unroll
  for (int kc = 0; kc < 8; ++kc) {
    union { v16h v; v8h h[2]; } u;
    const v8h* p = reinterpret_cast<const v8h*>(&yL[(tw + ln) * 264 + kc * 32 + hi * 8]);
    u.h[0] = p[0]; u.h[1] = p[2];
    a[kc] = u.v;
  }
  // logits -> out[b][o][t] (contiguous 8 floats per lane)
  float* ob = out + (size_t)b * 256 * T_;
#pragma unroll
  for (int nt = 0; nt < 16; ++nt) {
    int o = nt * 16 + ln;
    v8f c;
#pragma unroll
    for (int r = 0; r < 8; ++r) c[r] = lob[o];
#pragma unroll
    for (int kc = 0; kc < 8; ++kc) c = wmma_f16(a[kc], loadB_pk(loP, kc, 16, nt, lane), c);
    *reinterpret_cast<v8f*>(ob + (size_t)o * T_ + tb) = c;
  }
}

// ---------------------------------------------------------------------------
// Host side
// ---------------------------------------------------------------------------
struct Net {
  const float *x, *c;
  const float *first_w, *first_b, *skipc_w, *skipc_b, *condF_w, *condF_b;
  const float *fc_w, *fc_b, *logits_w, *logits_b;
  const float *causal_w[NL_], *causal_b[NL_], *cond_w[NL_], *cond_b[NL_];
  const float *res_w[NL_], *res_b[NL_], *skip_w[NL_], *skip_b[NL_];
};

static void resolve(void* const* d, const int* sz, Net& N) {
  auto F = [&](int i) { return (const float*)d[i]; };
  if (sz[0] == B_ * T_) {
    // insertion order: x, c, first_w, first_b, skipc_w, skipc_b, cond_w, cond_b,
    // fc_w, fc_b, logits_w, logits_b, layers[20]{causal_w,causal_b,cond_w,
    // cond_b,res_w,res_b,skip_w,skip_b}
    N.x = F(0); N.c = F(1);
    N.first_w = F(2);  N.first_b = F(3);
    N.skipc_w = F(4);  N.skipc_b = F(5);
    N.condF_w = F(6);  N.condF_b = F(7);
    N.fc_w = F(8);     N.fc_b = F(9);
    N.logits_w = F(10); N.logits_b = F(11);
    for (int l = 0; l < NL_; ++l) {
      int k = 12 + l * 8;
      N.causal_w[l] = F(k + 0); N.causal_b[l] = F(k + 1);
      N.cond_w[l]   = F(k + 2); N.cond_b[l]   = F(k + 3);
      N.res_w[l]    = F(k + 4); N.res_b[l]    = F(k + 5);
      N.skip_w[l]   = F(k + 6); N.skip_b[l]   = F(k + 7);
    }
  } else {
    // jax pytree sorted-key order: c, cond_b, cond_w, fc_b, fc_w, first_b,
    // first_w, layers[20]{causal_b,causal_w,cond_b,cond_w,res_b,res_w,skip_b,
    // skip_w}, logits_b, logits_w, skipc_b, skipc_w, x
    N.c = F(0);
    N.condF_b = F(1); N.condF_w = F(2);
    N.fc_b = F(3);    N.fc_w = F(4);
    N.first_b = F(5); N.first_w = F(6);
    for (int l = 0; l < NL_; ++l) {
      int k = 7 + l * 8;
      N.causal_b[l] = F(k + 0); N.causal_w[l] = F(k + 1);
      N.cond_b[l]   = F(k + 2); N.cond_w[l]   = F(k + 3);
      N.res_b[l]    = F(k + 4); N.res_w[l]    = F(k + 5);
      N.skip_b[l]   = F(k + 6); N.skip_w[l]   = F(k + 7);
    }
    N.logits_b = F(167); N.logits_w = F(168);
    N.skipc_b = F(169);  N.skipc_w = F(170);
    N.x = F(171);
  }
}

// workspace offsets (bytes, 256-aligned)
static const size_t OFF_RES_A = 0;
static const size_t OFF_RES_B = OFF_RES_A + (size_t)B_ * T_ * RC_ * 2;     //  16 MB
static const size_t OFF_SKIP  = OFF_RES_B + (size_t)B_ * T_ * RC_ * 2;     //  16 MB
static const size_t OFF_CONDP = OFF_SKIP  + (size_t)B_ * T_ * SC_ * 4;     // 128 MB
static const size_t OFF_CONDF = OFF_CONDP + (size_t)NL_ * B_ * 128 * 128 * 4;
static const size_t OFF_WPACK = OFF_CONDF + (size_t)B_ * 128 * 256 * 4;
static const size_t OFF_FCP   = OFF_WPACK + (size_t)NL_ * WL_HALF_ * 2;
static const size_t OFF_LOGP  = OFF_FCP + (size_t)256 * 256 * 2;

extern "C" void kernel_launch(void* const* d_in, const int* in_sizes, int n_in,
                              void* d_out, int out_size, void* d_ws, size_t ws_size,
                              hipStream_t stream) {
  (void)n_in; (void)out_size; (void)ws_size;
  Net N; resolve(d_in, in_sizes, N);

  char* ws = (char*)d_ws;
  _Float16* resA  = (_Float16*)(ws + OFF_RES_A);
  _Float16* resB  = (_Float16*)(ws + OFF_RES_B);
  float*    skipB = (float*)(ws + OFF_SKIP);
  float*    condp = (float*)(ws + OFF_CONDP);
  float*    condf = (float*)(ws + OFF_CONDF);
  _Float16* wpack = (_Float16*)(ws + OFF_WPACK);
  _Float16* fcP   = (_Float16*)(ws + OFF_FCP);
  _Float16* loP   = (_Float16*)(ws + OFF_LOGP);

  // 1) fragment-swizzle all weights to f16
  for (int l = 0; l < NL_; ++l) {
    _Float16* wl = wpack + (size_t)l * WL_HALF_;
    pack_weight<<<(8192 + 255) / 256, 256, 0, stream>>>(N.causal_w[l], wl + WC0_OFF_, 128, 64, 0, 2);
    pack_weight<<<(8192 + 255) / 256, 256, 0, stream>>>(N.causal_w[l], wl + WC1_OFF_, 128, 64, 1, 2);
    pack_weight<<<(4096 + 255) / 256, 256, 0, stream>>>(N.res_w[l],    wl + WR_OFF_,   64, 64, 0, 1);
    pack_weight<<<(16384 + 255) / 256, 256, 0, stream>>>(N.skip_w[l],  wl + WS_OFF_,  256, 64, 0, 1);
  }
  pack_weight<<<(65536 + 255) / 256, 256, 0, stream>>>(N.fc_w,     fcP, 256, 256, 0, 1);
  pack_weight<<<(65536 + 255) / 256, 256, 0, stream>>>(N.logits_w, loP, 256, 256, 0, 1);

  // 2) cond projections (128 distinct frames)
  for (int l = 0; l < NL_; ++l)
    cond_proj<<<(B_ * 128 * 128 + 255) / 256, 256, 0, stream>>>(
        N.c, N.cond_w[l], N.cond_b[l], condp + (size_t)l * B_ * 128 * 128, 128);
  cond_proj<<<(B_ * 128 * 256 + 255) / 256, 256, 0, stream>>>(
      N.c, N.condF_w, N.condF_b, condf, 256);

  // 3) first conv + skip init
  first_conv<<<(B_ * T_ + 255) / 256, 256, 0, stream>>>(N.x, N.first_w, N.first_b, resA);
  skip_init<<<dim3(T_ / 64, B_), 256, 0, stream>>>(resA, N.skipc_w, N.skipc_b, skipB);

  // 4) 20 fused layers, ping-pong residual
  _Float16 *rin = resA, *rout = resB;
  for (int l = 0; l < NL_; ++l) {
    int dil = 1 << (l % 10);
    wavenet_layer<<<dim3(T_ / 128, B_), 256, 0, stream>>>(
        rin, rout, skipB, wpack + (size_t)l * WL_HALF_,
        N.causal_b[l], N.res_b[l], N.skip_b[l],
        condp + (size_t)l * B_ * 128 * 128, dil);
    _Float16* t = rin; rin = rout; rout = t;
  }

  // 5) head
  wavenet_head<<<dim3(T_ / 64, B_), 128, 0, stream>>>(
      skipB, fcP, loP, N.fc_b, N.logits_b, condf, (float*)d_out);
}
